// TriAttnMHAWrapper_83708912599790
// MI455X (gfx1250) — compile-verified, matrix operand order checked by reading, not run on hardware
//
#include <hip/hip_runtime.h>

// ---------------------------------------------------------------------------
// Triangle attention (N=320, C_Z=128, H=4, C_H=32) on CDNA5 via bf16 WMMA.
// ---------------------------------------------------------------------------

typedef __attribute__((ext_vector_type(16))) __bf16 v16bf;
typedef __attribute__((ext_vector_type(8)))  __bf16 v8bf;
typedef __attribute__((ext_vector_type(8)))  float  v8f;

#define NSEQ 320
#define CZ   128
#define NH   4
#define CH   32
#define ROWS (NSEQ * NSEQ)   // 102400

// fp32 -> bf16 via native convert (v_cvt_pk_bf16_f32 on gfx1250)
__device__ inline unsigned short f32_to_bf16(float f) {
  __bf16 h = (__bf16)f;
  return __builtin_bit_cast(unsigned short, h);
}
__device__ inline float bf16u_to_f32(unsigned short h) {
  unsigned int x = ((unsigned int)h) << 16;
  return __builtin_bit_cast(float, x);
}

// Load a 16-element bf16 fragment as two b128 loads (A or B operand).
// klo/khi are element offsets: half*8 and 16+half*8 (ISA 16-bit operand striping).
__device__ inline v16bf load_frag16(const unsigned short* p, int klo, int khi) {
  v8bf lo = *(const v8bf*)(p + klo);
  v8bf hi = *(const v8bf*)(p + khi);
  v16bf r;
#pragma unroll
  for (int i = 0; i < 8; ++i) { r[i] = lo[i]; r[8 + i] = hi[i]; }
  return r;
}

__device__ inline v8f wmma_bf16(v16bf a, v16bf b, v8f c) {
  return __builtin_amdgcn_wmma_f32_16x16x32_bf16(false, a, false, b, (short)0, c,
                                                 false, false);
}

// ---------------------------------------------------------------------------
// Kernel 1: convert the five 128x128 fp32 weights to bf16, transposed
// (wt[n][k]) so WMMA B-fragments are contiguous b128 loads. q-scale folded in.
// ---------------------------------------------------------------------------
__global__ void __launch_bounds__(256)
wconv_kernel(const float* wq, const float* wk, const float* wv,
             const float* wg, const float* wo, unsigned short* wt) {
  int w = blockIdx.x;  // 0..4
  const float* src = (w == 0) ? wq : (w == 1) ? wk : (w == 2) ? wv
                   : (w == 3) ? wg : wo;
  float scale = (w == 0) ? 0.17677669529663687f : 1.0f;  // 1/sqrt(32) for q
  unsigned short* dst = wt + w * CZ * CZ;
  for (int e = threadIdx.x; e < CZ * CZ; e += blockDim.x) {
    int r = e >> 7, c = e & 127;
    dst[c * CZ + r] = f32_to_bf16(src[e] * scale);
  }
}

// ---------------------------------------------------------------------------
// Kernel 2: fused q/k/v/gate projection. 64 rows per block, x converted to
// bf16 in LDS. wsel unrolled at compile time; per-wave tile order keeps the
// column tile fixed so B-fragments stay hoisted in VGPRs across row tiles.
// ---------------------------------------------------------------------------
__global__ void __launch_bounds__(256)
proj_kernel(const float* __restrict__ x, const unsigned short* __restrict__ wt,
            const float* __restrict__ bg, unsigned short* __restrict__ qb,
            unsigned short* __restrict__ kb, unsigned short* __restrict__ vb,
            unsigned short* __restrict__ gb) {
  __shared__ __align__(16) unsigned short Xs[64 * CZ];  // 16 KB bf16 tile
  int row0 = blockIdx.x * 64;

  // Prefetch next block's x tile while we work on this one.
  if (blockIdx.x + 1 < gridDim.x)
    __builtin_prefetch(x + (size_t)(row0 + 64) * CZ + threadIdx.x * 32, 0, 0);

  // Stage 64x128 fp32 tile -> bf16 LDS (tile is fully contiguous in memory).
  const float4* xsrc = (const float4*)(x + (size_t)row0 * CZ);
#pragma unroll
  for (int i = 0; i < 8; ++i) {
    int idx = threadIdx.x + i * 256;  // 2048 float4 total
    float4 f = xsrc[idx];
    unsigned short* d = &Xs[idx * 4];
    d[0] = f32_to_bf16(f.x); d[1] = f32_to_bf16(f.y);
    d[2] = f32_to_bf16(f.z); d[3] = f32_to_bf16(f.w);
  }
  __syncthreads();

  int lane = threadIdx.x & 31, wave = threadIdx.x >> 5;
  int half = lane >> 4, lm = lane & 15;
  int klo0 = half * 8, khi0 = 16 + half * 8;
  int ct = wave;                  // each wave owns one 16-col tile
  int col = ct * 16 + lm;

  unsigned short* const outs[4] = {qb, kb, vb, gb};
#pragma unroll
  for (int wsel = 0; wsel < 4; ++wsel) {
    unsigned short* outp = outs[wsel];
    const unsigned short* W = wt + wsel * CZ * CZ + col * CZ;
    float badd = (wsel == 3) ? bg[col] : 0.0f;
    for (int rt = 0; rt < 4; ++rt) {   // B-fragments reused across row tiles
      v8f acc = {};
#pragma unroll
      for (int kk = 0; kk < 4; ++kk) {
        int k0 = kk * 32;
        v16bf a = load_frag16(&Xs[(rt * 16 + lm) * CZ + k0], klo0, khi0);
        v16bf b = load_frag16(W + k0, klo0, khi0);
        acc = wmma_bf16(a, b, acc);
      }
      // Base pointer + compile-time row offsets -> instruction-offset stores.
      unsigned short* p =
          outp + (size_t)(row0 + rt * 16 + half * 8) * CZ + col;
#pragma unroll
      for (int j = 0; j < 8; ++j) {
        float val = acc[j];
        if (wsel == 3) val = 1.0f / (1.0f + __expf(-(val + badd)));  // sigmoid
        p[j * CZ] = f32_to_bf16(val);
      }
    }
  }
}

// ---------------------------------------------------------------------------
// Kernel 3: per-(n,h) flash attention. K staged row-major in LDS, V staged
// transposed; per-wave online softmax over 10 chunks of 32 keys. Triangle
// bias is loaded directly into the WMMA C-fragment (layout matches).
// ---------------------------------------------------------------------------
__global__ void __launch_bounds__(256)
attn_kernel(const unsigned short* __restrict__ qg,
            const unsigned short* __restrict__ kg,
            const unsigned short* __restrict__ vg,
            const unsigned short* __restrict__ gg,
            const float* __restrict__ bias, unsigned short* __restrict__ og) {
  int n = blockIdx.x >> 2;
  int h = blockIdx.x & 3;

  __shared__ __align__(16) unsigned short Ks[NSEQ * CH];   // 20 KB [key][c]
  __shared__ __align__(16) unsigned short Vt[CH * NSEQ];   // 20 KB [c][key]
  __shared__ __align__(16) unsigned short Ps[8][16 * 32];  //  8 KB P scratch

  // K copy (dword granularity, coalesced reads).
  const unsigned int* ksrc = (const unsigned int*)kg;
  unsigned int* kdst = (unsigned int*)Ks;
  for (int i = threadIdx.x; i < NSEQ * CH / 2; i += 256) {
    int m = i >> 4, cp = i & 15;
    kdst[i] = ksrc[((size_t)n * NSEQ + m) * (CZ / 2) + h * 16 + cp];
  }
  // V transpose copy.
  for (int i = threadIdx.x; i < NSEQ * CH; i += 256) {
    int m = i >> 5, c = i & 31;
    Vt[c * NSEQ + m] = vg[((size_t)n * NSEQ + m) * CZ + h * CH + c];
  }
  __syncthreads();

  int lane = threadIdx.x & 31, wave = threadIdx.x >> 5;
  int half = lane >> 4, lm = lane & 15;
  int klo0 = half * 8, khi0 = 16 + half * 8;
  const float* bh = bias + (size_t)h * NSEQ * NSEQ;

  for (int qt = wave; qt < 20; qt += 8) {
    int q0 = qt * 16;
    // Q A-fragment straight from global (contiguous head-dim pairs).
    v16bf aq = load_frag16(qg + ((size_t)n * NSEQ + q0 + lm) * CZ + h * CH,
                           klo0, khi0);
    v8f o0 = {}, o1 = {};
    float mrun[8], lrun[8];
#pragma unroll
    for (int j = 0; j < 8; ++j) { mrun[j] = -1e30f; lrun[j] = 0.0f; }

    const float* bbase = bh + (size_t)(q0 + half * 8) * NSEQ + lm;
    unsigned short* psw = &Ps[wave][(half * 8) * 32 + lm];

    for (int ch = 0; ch < 10; ++ch) {
      int kb0 = ch * 32;
      // Triangle bias rides the WMMA accumulator (C layout == bias tile).
      const float* bp = bbase + kb0;
      v8f s0, s1;
#pragma unroll
      for (int j = 0; j < 8; ++j) {
        s0[j] = bp[j * NSEQ];
        s1[j] = bp[j * NSEQ + 16];
      }
      v16bf b0 = load_frag16(&Ks[(kb0 + lm) * CH], klo0, khi0);
      v16bf b1 = load_frag16(&Ks[(kb0 + 16 + lm) * CH], klo0, khi0);
      s0 = wmma_bf16(aq, b0, s0);
      s1 = wmma_bf16(aq, b1, s1);

      float rmax[8];
#pragma unroll
      for (int j = 0; j < 8; ++j) {
        float mx = fmaxf(s0[j], s1[j]);
#pragma unroll
        for (int d = 1; d < 16; d <<= 1) mx = fmaxf(mx, __shfl_xor(mx, d, 32));
        rmax[j] = mx;  // row max over this chunk's 32 key columns
      }
#pragma unroll
      for (int j = 0; j < 8; ++j) {
        float mn   = fmaxf(mrun[j], rmax[j]);
        float corr = __expf(mrun[j] - mn);
        float p0   = __expf(s0[j] - mn);
        float p1   = __expf(s1[j] - mn);
        float sum  = p0 + p1;
#pragma unroll
        for (int d = 1; d < 16; d <<= 1) sum += __shfl_xor(sum, d, 32);
        lrun[j] = lrun[j] * corr + sum;
        mrun[j] = mn;
        o0[j] *= corr; o1[j] *= corr;
        // Re-stripe P from C-layout to A-layout via per-wave LDS scratch.
        psw[j * 32]      = f32_to_bf16(p0);
        psw[j * 32 + 16] = f32_to_bf16(p1);
      }
      v16bf ap  = load_frag16(&Ps[wave][lm * 32], klo0, khi0);
      v16bf bv0 = load_frag16(&Vt[lm * NSEQ + kb0], klo0, khi0);
      v16bf bv1 = load_frag16(&Vt[(16 + lm) * NSEQ + kb0], klo0, khi0);
      o0 = wmma_bf16(ap, bv0, o0);
      o1 = wmma_bf16(ap, bv1, o1);
    }

    // Normalize, sigmoid-gate, write bf16 (base + immediate offsets).
    size_t rbase = ((size_t)n * NSEQ + q0 + half * 8) * CZ + h * CH + lm;
    const unsigned short* gp = gg + rbase;
    unsigned short* op = og + rbase;
#pragma unroll
    for (int j = 0; j < 8; ++j) {
      float inv = 1.0f / lrun[j];
      float g0 = bf16u_to_f32(gp[j * CZ]);
      float g1 = bf16u_to_f32(gp[j * CZ + 16]);
      op[j * CZ]      = f32_to_bf16(o0[j] * inv * g0);
      op[j * CZ + 16] = f32_to_bf16(o1[j] * inv * g1);
    }
  }
}

// ---------------------------------------------------------------------------
// Kernel 4: out = gated_o @ wo + bo  (bf16 WMMA, fp32 out)
// ---------------------------------------------------------------------------
__global__ void __launch_bounds__(256)
out_kernel(const unsigned short* __restrict__ og,
           const unsigned short* __restrict__ wot,
           const float* __restrict__ bo, float* __restrict__ out) {
  __shared__ __align__(16) unsigned short Xs[64 * CZ];
  int row0 = blockIdx.x * 64;

  if (blockIdx.x + 1 < gridDim.x)
    __builtin_prefetch(og + (size_t)(row0 + 64) * CZ + threadIdx.x * 16, 0, 0);

  const uint4* src = (const uint4*)(og + (size_t)row0 * CZ);
  uint4* dst = (uint4*)Xs;
#pragma unroll
  for (int i = 0; i < 4; ++i) dst[threadIdx.x + i * 256] = src[threadIdx.x + i * 256];
  __syncthreads();

  int lane = threadIdx.x & 31, wave = threadIdx.x >> 5;
  int half = lane >> 4, lm = lane & 15;
  int klo0 = half * 8, khi0 = 16 + half * 8;
  int ct = wave;
  int col = ct * 16 + lm;
  const unsigned short* W = wot + col * CZ;
  float bb = bo[col];

  for (int rt = 0; rt < 4; ++rt) {     // B-fragments reused across row tiles
    v8f acc = {};
#pragma unroll
    for (int kk = 0; kk < 4; ++kk) {
      v16bf a = load_frag16(&Xs[(rt * 16 + lm) * CZ + kk * 32], klo0, khi0);
      v16bf b = load_frag16(W + kk * 32, klo0, khi0);
      acc = wmma_bf16(a, b, acc);
    }
    float* p = out + (size_t)(row0 + rt * 16 + half * 8) * CZ + col;
#pragma unroll
    for (int j = 0; j < 8; ++j) p[j * CZ] = acc[j] + bb;
  }
}

// ---------------------------------------------------------------------------
// Launch
// ---------------------------------------------------------------------------
extern "C" void kernel_launch(void* const* d_in, const int* in_sizes, int n_in,
                              void* d_out, int out_size, void* d_ws, size_t ws_size,
                              hipStream_t stream) {
  const float* x    = (const float*)d_in[0];
  const float* bias = (const float*)d_in[1];
  const float* wq   = (const float*)d_in[2];
  const float* wk   = (const float*)d_in[3];
  const float* wv   = (const float*)d_in[4];
  const float* wg   = (const float*)d_in[5];
  const float* bg   = (const float*)d_in[6];
  const float* wo   = (const float*)d_in[7];
  const float* bo   = (const float*)d_in[8];

  // Workspace layout (bf16/u16 elements): 5 transposed weights, then
  // q, k, v, gate, gated-o (each ROWS x 128 bf16).  Total ~131 MB.
  unsigned short* ws = (unsigned short*)d_ws;
  unsigned short* wt  = ws;                       // 5 * 16384
  unsigned short* qb  = wt + 5 * CZ * CZ;
  unsigned short* kb  = qb + (size_t)ROWS * CZ;
  unsigned short* vb  = kb + (size_t)ROWS * CZ;
  unsigned short* gb  = vb + (size_t)ROWS * CZ;
  unsigned short* ogb = gb + (size_t)ROWS * CZ;

  wconv_kernel<<<5, 256, 0, stream>>>(wq, wk, wv, wg, wo, wt);
  proj_kernel<<<ROWS / 64, 256, 0, stream>>>(x, wt, bg, qb, kb, vb, gb);
  attn_kernel<<<NSEQ * NH, 256, 0, stream>>>(qb, kb, vb, gb, bias, ogb);
  out_kernel<<<ROWS / 64, 256, 0, stream>>>(ogb, wt + 4 * CZ * CZ, bo,
                                            (float*)d_out);
}
